// DynamicHeadTime_63634235458242
// MI455X (gfx1250) — matrix operand order repaired and checked by reading, not compiled, not run on hardware
//
#include <hip/hip_runtime.h>
#include <hip/hip_bf16.h>
#include <math.h>

#define LL   6
#define NN   4
#define NR_  100
#define NB   400
#define TT   256
#define DD   256
#define NHH  8
#define HD   32
#define FF_  2048
#define RES_ 14
#define NC_  80
#define DYN_ 64
#define NP2  32768   // 2 * D * DYN

typedef __attribute__((ext_vector_type(16))) _Float16 v16h;
typedef __attribute__((ext_vector_type(8)))  float    v8f;

// ---------------------------------------------------------------------------
// Fragment loaders (fp32 memory -> f16 WMMA fragments)
// A fragment: lane supplies row m; elements = K+{0..7} (lo half-lane) and
// K+{16..23}, shifted by 8 for hi half-lanes -> base pointer pre-offset by
// (half<<3). B fragment: lane supplies column n; same K order as A,
// base pre-offset by (half<<3).
// ---------------------------------------------------------------------------
__device__ inline v16h load_frag_a(const float* __restrict__ p)
{
    float4 a0 = *(const float4*)(p);
    float4 a1 = *(const float4*)(p + 4);
    float4 a2 = *(const float4*)(p + 16);
    float4 a3 = *(const float4*)(p + 20);
    v16h v;
    v[0]=(_Float16)a0.x;  v[1]=(_Float16)a0.y;  v[2]=(_Float16)a0.z;  v[3]=(_Float16)a0.w;
    v[4]=(_Float16)a1.x;  v[5]=(_Float16)a1.y;  v[6]=(_Float16)a1.z;  v[7]=(_Float16)a1.w;
    v[8]=(_Float16)a2.x;  v[9]=(_Float16)a2.y;  v[10]=(_Float16)a2.z; v[11]=(_Float16)a2.w;
    v[12]=(_Float16)a3.x; v[13]=(_Float16)a3.y; v[14]=(_Float16)a3.z; v[15]=(_Float16)a3.w;
    return v;
}

__device__ inline v16h load_frag_b(const float* __restrict__ p)
{
    float4 w0 = *(const float4*)(p);
    float4 w1 = *(const float4*)(p + 4);
    float4 w2 = *(const float4*)(p + 16);
    float4 w3 = *(const float4*)(p + 20);
    v16h v;
    v[0]=(_Float16)w0.x;  v[1]=(_Float16)w0.y;  v[2]=(_Float16)w0.z;  v[3]=(_Float16)w0.w;
    v[4]=(_Float16)w1.x;  v[5]=(_Float16)w1.y;  v[6]=(_Float16)w1.z;  v[7]=(_Float16)w1.w;
    v[8]=(_Float16)w2.x;  v[9]=(_Float16)w2.y;  v[10]=(_Float16)w2.z; v[11]=(_Float16)w2.w;
    v[12]=(_Float16)w3.x; v[13]=(_Float16)w3.y; v[14]=(_Float16)w3.z; v[15]=(_Float16)w3.w;
    return v;
}

// ---------------------------------------------------------------------------
// Generic GEMM: C[M,N] = act( A[M,K] @ W[N,K]^T + bias )
// One wave per 32x32 output tile (2x2 register-blocked 16x16 WMMA tiles,
// 4x v_wmma_f32_16x16x32_f16 per K step). Out-of-range rows/cols are handled
// by clamping load addresses (branch-free inner loop) and masking stores.
// ---------------------------------------------------------------------------
__global__ __launch_bounds__(32)
void gemm_f16_wmma(const float* __restrict__ A, const float* __restrict__ W,
                   const float* __restrict__ bias, float* __restrict__ C,
                   int M, int N, int K, int relu)
{
    const int lane = threadIdx.x & 31;
    const int half = lane >> 4;
    const int l16  = lane & 15;
    const int n0 = blockIdx.x * 32;
    const int m0 = blockIdx.y * 32;

    // Clamped row pointers: garbage rows only feed masked-out C entries.
    const float* a0p = A + (size_t)min(m0 + l16,      M - 1) * K + (half << 3);
    const float* a1p = A + (size_t)min(m0 + 16 + l16, M - 1) * K + (half << 3);
    const float* b0p = W + (size_t)min(n0 + l16,      N - 1) * K + (half << 3);
    const float* b1p = W + (size_t)min(n0 + 16 + l16, N - 1) * K + (half << 3);

    v8f acc00 = {}, acc01 = {}, acc10 = {}, acc11 = {};
    for (int kk = 0; kk < K; kk += 32) {
        v16h av0 = load_frag_a(a0p + kk);
        v16h av1 = load_frag_a(a1p + kk);
        v16h bv0 = load_frag_b(b0p + kk);
        v16h bv1 = load_frag_b(b1p + kk);
        if (kk + 32 < K) {
            __builtin_prefetch(b0p + kk + 32, 0, 1);
            __builtin_prefetch(b1p + kk + 32, 0, 1);
        }
        acc00 = __builtin_amdgcn_wmma_f32_16x16x32_f16(false, av0, false, bv0,
                                                       (short)0, acc00, false, false);
        acc01 = __builtin_amdgcn_wmma_f32_16x16x32_f16(false, av0, false, bv1,
                                                       (short)0, acc01, false, false);
        acc10 = __builtin_amdgcn_wmma_f32_16x16x32_f16(false, av1, false, bv0,
                                                       (short)0, acc10, false, false);
        acc11 = __builtin_amdgcn_wmma_f32_16x16x32_f16(false, av1, false, bv1,
                                                       (short)0, acc11, false, false);
    }

    // D layout: lane half h, n = nb + l16; VGPR r -> M = mb + r + 8*h
    #pragma unroll
    for (int tile = 0; tile < 4; ++tile) {
        const v8f& acc = (tile == 0) ? acc00 : (tile == 1) ? acc01
                        : (tile == 2) ? acc10 : acc11;
        int mb = m0 + ((tile >> 1) << 4);
        int nb = n0 + ((tile & 1) << 4);
        int n = nb + l16;
        if (n < N) {
            float bv = bias ? bias[n] : 0.f;
            #pragma unroll
            for (int r = 0; r < 8; ++r) {
                int m = mb + r + (half << 3);
                if (m < M) {
                    float v = acc[r] + bv;
                    if (relu) v = fmaxf(v, 0.f);
                    C[(size_t)m * N + n] = v;
                }
            }
        }
    }
}

// ---------------------------------------------------------------------------
// Row LayerNorm: Y = act( LN(X [+ Res]) * g + b ), cols <= 256, one block/row
// ---------------------------------------------------------------------------
__global__ __launch_bounds__(256)
void ln_kernel(const float* __restrict__ X, const float* __restrict__ Res,
               const float* __restrict__ g, const float* __restrict__ b,
               float* __restrict__ Y, int cols, int relu)
{
    __shared__ float red[256];
    const int row = blockIdx.x;
    const int t = threadIdx.x;
    float x = 0.f;
    if (t < cols) {
        x = X[(size_t)row * cols + t];
        if (Res) x += Res[(size_t)row * cols + t];
    }
    red[t] = (t < cols) ? x : 0.f;
    __syncthreads();
    for (int s = 128; s > 0; s >>= 1) { if (t < s) red[t] += red[t + s]; __syncthreads(); }
    float mean = red[0] / cols;
    __syncthreads();
    float d = (t < cols) ? (x - mean) : 0.f;
    red[t] = d * d;
    __syncthreads();
    for (int s = 128; s > 0; s >>= 1) { if (t < s) red[t] += red[t + s]; __syncthreads(); }
    float inv = rsqrtf(red[0] / cols + 1e-5f);
    if (t < cols) {
        float y = (x - mean) * inv * g[t] + b[t];
        if (relu) y = fmaxf(y, 0.f);
        Y[(size_t)row * cols + t] = y;
    }
}

// ---------------------------------------------------------------------------
// Time embedding: ste = silu( MLP(sincos(t)) )   (N x 1024)
// ---------------------------------------------------------------------------
__device__ inline float gelu_f(float x) {
    float x3 = x * x * x;
    return 0.5f * x * (1.f + tanhf(0.7978845608028654f * (x + 0.044715f * x3)));
}

__global__ __launch_bounds__(256)
void te_kernel(const float* __restrict__ tin,
               const float* __restrict__ w1, const float* __restrict__ b1,
               const float* __restrict__ w2, const float* __restrict__ b2,
               float* __restrict__ ste)
{
    __shared__ float temb[NN * DD];
    __shared__ float hbuf[NN * 4 * DD];
    const int t = threadIdx.x;
    const float c = -logf(10000.f) / 127.f;
    for (int idx = t; idx < NN * DD; idx += 256) {
        int n = idx / DD, d = idx % DD;
        float v;
        if (d < 128) v = sinf(tin[n] * expf(d * c));
        else         v = cosf(tin[n] * expf((d - 128) * c));
        temb[idx] = v;
    }
    __syncthreads();
    for (int idx = t; idx < NN * 4 * DD; idx += 256) {
        int n = idx / (4 * DD), o = idx % (4 * DD);
        float s = b1[o];
        const float* wr = w1 + (size_t)o * DD;
        for (int k = 0; k < DD; ++k) s += temb[n * DD + k] * wr[k];
        hbuf[idx] = gelu_f(s);
    }
    __syncthreads();
    for (int idx = t; idx < NN * 4 * DD; idx += 256) {
        int n = idx / (4 * DD), o = idx % (4 * DD);
        float s = b2[o];
        const float* wr = w2 + (size_t)o * 4 * DD;
        for (int k = 0; k < 4 * DD; ++k) s += hbuf[n * 4 * DD + k] * wr[k];
        ste[idx] = s * (1.f / (1.f + expf(-s)));   // silu
    }
}

// ---------------------------------------------------------------------------
// ROI bilinear sampling: feat[b][r][d], optionally pro[b][d] = mean_r
// ---------------------------------------------------------------------------
__global__ __launch_bounds__(256)
void roi_kernel(const float* __restrict__ features, const float* __restrict__ boxes,
                float* __restrict__ feat, float* __restrict__ pro, int write_pro)
{
    __shared__ float sroi[RES_ * DD];
    const int b = blockIdx.x;
    const int n = b / NR_;
    const float s = boxes[b * 2], e = boxes[b * 2 + 1];
    const int t = threadIdx.x;
    for (int idx = t; idx < RES_ * DD; idx += 256) {
        int r = idx / DD, d = idx % DD;
        float bin = (r + 0.5f) / (float)RES_;
        float pos = s + bin * (e - s);
        pos = fminf(fmaxf(pos, 0.f), (float)(TT - 1));
        int lo = (int)floorf(pos);
        int hi = min(lo + 1, TT - 1);
        float w = pos - (float)lo;
        const float* f = features + ((size_t)n * DD + d) * TT;
        float v = f[lo] * (1.f - w) + f[hi] * w;
        sroi[idx] = v;
        feat[(size_t)b * RES_ * DD + idx] = v;
    }
    if (write_pro) {
        __syncthreads();
        for (int d = t; d < DD; d += 256) {
            float acc = 0.f;
            for (int r = 0; r < RES_; ++r) acc += sroi[r * DD + d];
            pro[(size_t)b * DD + d] = acc / (float)RES_;
        }
    }
}

// ---------------------------------------------------------------------------
// Per-(batch,head) softmax attention over 100 queries / 100 keys / 32 dims
// ---------------------------------------------------------------------------
__global__ __launch_bounds__(128)
void attn_kernel(const float* __restrict__ qkv, float* __restrict__ att)
{
    __shared__ float kb[NR_ * HD];
    __shared__ float vb[NR_ * HD];
    const int bi = blockIdx.x / NHH;
    const int h  = blockIdx.x % NHH;
    const int t  = threadIdx.x;
    for (int idx = t; idx < NR_ * HD; idx += 128) {
        int kr = idx / HD, d = idx % HD;
        const float* row = qkv + (size_t)(bi * NR_ + kr) * (3 * DD);
        kb[idx] = row[DD + h * HD + d];
        vb[idx] = row[2 * DD + h * HD + d];
    }
    __syncthreads();
    if (t < NR_) {
        float q[HD];
        const float* row = qkv + (size_t)(bi * NR_ + t) * (3 * DD) + h * HD;
        for (int d = 0; d < HD; ++d) q[d] = row[d];
        const float sc = 0.17677669529663687f;  // 1/sqrt(32)
        float mx = -1e30f;
        for (int kr = 0; kr < NR_; ++kr) {
            float s = 0.f;
            for (int d = 0; d < HD; ++d) s += q[d] * kb[kr * HD + d];
            mx = fmaxf(mx, s * sc);
        }
        float den = 0.f;
        for (int kr = 0; kr < NR_; ++kr) {
            float s = 0.f;
            for (int d = 0; d < HD; ++d) s += q[d] * kb[kr * HD + d];
            den += expf(s * sc - mx);
        }
        float acc[HD];
        for (int d = 0; d < HD; ++d) acc[d] = 0.f;
        for (int kr = 0; kr < NR_; ++kr) {
            float s = 0.f;
            for (int d = 0; d < HD; ++d) s += q[d] * kb[kr * HD + d];
            float w = expf(s * sc - mx) / den;
            for (int d = 0; d < HD; ++d) acc[d] += w * vb[kr * HD + d];
        }
        float* o = att + (size_t)(bi * NR_ + t) * DD + h * HD;
        for (int d = 0; d < HD; ++d) o[d] = acc[d];
    }
}

// ---------------------------------------------------------------------------
// Per-box dynamic conv: relu(LN(feat@p1)) @ p2 -> relu(LN) ; out (Nb,14,256)
// ---------------------------------------------------------------------------
__global__ __launch_bounds__(256)
void dynconv_kernel(const float* __restrict__ feat, const float* __restrict__ params,
                    const float* __restrict__ g1, const float* __restrict__ b1,
                    const float* __restrict__ g2, const float* __restrict__ b2,
                    float* __restrict__ out)
{
    __shared__ float sf[RES_ * DD];
    __shared__ float t1[RES_ * DYN_];
    __shared__ float t2[RES_ * DD];
    __shared__ float rmean[RES_], rinv[RES_];
    const int b = blockIdx.x;
    const int t = threadIdx.x;
    const float* pr = params + (size_t)b * NP2;
    for (int idx = t; idx < RES_ * DD; idx += 256)
        sf[idx] = feat[(size_t)b * RES_ * DD + idx];
    __syncthreads();
    for (int idx = t; idx < RES_ * DYN_; idx += 256) {
        int r = idx / DYN_, e2 = idx % DYN_;
        float s = 0.f;
        for (int d = 0; d < DD; ++d) s += sf[r * DD + d] * pr[d * DYN_ + e2];
        t1[idx] = s;
    }
    __syncthreads();
    if (t < RES_) {
        float m = 0.f;
        for (int e2 = 0; e2 < DYN_; ++e2) m += t1[t * DYN_ + e2];
        m /= DYN_;
        float v = 0.f;
        for (int e2 = 0; e2 < DYN_; ++e2) { float d0 = t1[t * DYN_ + e2] - m; v += d0 * d0; }
        rmean[t] = m; rinv[t] = rsqrtf(v / DYN_ + 1e-5f);
    }
    __syncthreads();
    for (int idx = t; idx < RES_ * DYN_; idx += 256) {
        int r = idx / DYN_, e2 = idx % DYN_;
        t1[idx] = fmaxf((t1[idx] - rmean[r]) * rinv[r] * g1[e2] + b1[e2], 0.f);
    }
    __syncthreads();
    for (int idx = t; idx < RES_ * DD; idx += 256) {
        int r = idx / DD, d = idx % DD;
        float s = 0.f;
        const float* p2 = pr + DD * DYN_;
        for (int e2 = 0; e2 < DYN_; ++e2) s += t1[r * DYN_ + e2] * p2[e2 * DD + d];
        t2[idx] = s;
    }
    __syncthreads();
    if (t < RES_) {
        float m = 0.f;
        for (int d = 0; d < DD; ++d) m += t2[t * DD + d];
        m /= DD;
        float v = 0.f;
        for (int d = 0; d < DD; ++d) { float d0 = t2[t * DD + d] - m; v += d0 * d0; }
        rmean[t] = m; rinv[t] = rsqrtf(v / DD + 1e-5f);
    }
    __syncthreads();
    for (int idx = t; idx < RES_ * DD; idx += 256) {
        int r = idx / DD, d = idx % DD;
        out[(size_t)b * RES_ * DD + idx] =
            fmaxf((t2[idx] - rmean[r]) * rinv[r] * g2[d] + b2[d], 0.f);
    }
}

// FiLM: fc = obj * (scale + 1) + shift
__global__ void film_kernel(const float* __restrict__ obj, const float* __restrict__ ss,
                            float* __restrict__ fc)
{
    int i = blockIdx.x * blockDim.x + threadIdx.x;
    if (i >= NB * DD) return;
    int bq = i / DD, d = i % DD;
    int n = bq / NR_;
    float sc = ss[n * 2 * DD + d];
    float sh = ss[n * 2 * DD + DD + d];
    fc[i] = obj[i] * (sc + 1.f) + sh;
}

// Box decode + update
__global__ void boxupdate_kernel(const float* __restrict__ dl, float* __restrict__ boxes,
                                 float* __restrict__ out_boxes)
{
    int b = blockIdx.x * blockDim.x + threadIdx.x;
    if (b >= NB) return;
    float s = boxes[b * 2], e = boxes[b * 2 + 1];
    float dur = e - s, ctr = s + 0.5f * dur;
    float pc = dl[b * 2] * 0.5f * dur + ctr;
    float pd = expf(fminf(dl[b * 2 + 1], 8.740336742730447f)) * dur;
    float p0 = pc - 0.5f * pd, p1 = pc + 0.5f * pd;
    out_boxes[b * 2] = p0; out_boxes[b * 2 + 1] = p1;
    boxes[b * 2] = p0;     boxes[b * 2 + 1] = p1;
}

// ---------------------------------------------------------------------------
extern "C" void kernel_launch(void* const* d_in, const int* in_sizes, int n_in,
                              void* d_out, int out_size, void* d_ws, size_t ws_size,
                              hipStream_t stream)
{
    (void)in_sizes; (void)n_in; (void)out_size; (void)ws_size;

    const float* features   = (const float*)d_in[0];
    const float* bboxes     = (const float*)d_in[1];
    const float* t_in       = (const float*)d_in[2];
    const float* tm_w1      = (const float*)d_in[3];
    const float* tm_b1      = (const float*)d_in[4];
    const float* tm_w2      = (const float*)d_in[5];
    const float* tm_b2      = (const float*)d_in[6];
    const float* attn_in_w  = (const float*)d_in[7];
    const float* attn_in_b  = (const float*)d_in[8];
    const float* attn_out_w = (const float*)d_in[9];
    const float* attn_out_b = (const float*)d_in[10];
    const float* ln1_g = (const float*)d_in[11];
    const float* ln1_b = (const float*)d_in[12];
    const float* ln2_g = (const float*)d_in[13];
    const float* ln2_b = (const float*)d_in[14];
    const float* ln3_g = (const float*)d_in[15];
    const float* ln3_b = (const float*)d_in[16];
    const float* dyn_w = (const float*)d_in[17];
    const float* dyn_b = (const float*)d_in[18];
    const float* dc_ln1_g = (const float*)d_in[19];
    const float* dc_ln1_b = (const float*)d_in[20];
    const float* dc_ln2_g = (const float*)d_in[21];
    const float* dc_ln2_b = (const float*)d_in[22];
    const float* dc_out_w = (const float*)d_in[23];
    const float* dc_out_b = (const float*)d_in[24];
    const float* dc_ln3_g = (const float*)d_in[25];
    const float* dc_ln3_b = (const float*)d_in[26];
    const float* ffn_w1 = (const float*)d_in[27];
    const float* ffn_b1 = (const float*)d_in[28];
    const float* ffn_w2 = (const float*)d_in[29];
    const float* ffn_b2 = (const float*)d_in[30];
    const float* tb_w = (const float*)d_in[31];
    const float* tb_b = (const float*)d_in[32];
    const float* cls_w = (const float*)d_in[33];
    const float* cls_ln_g = (const float*)d_in[34];
    const float* cls_ln_b = (const float*)d_in[35];
    const float* reg_w = (const float*)d_in[36];
    const float* reg_ln_g = (const float*)d_in[37];
    const float* reg_ln_b = (const float*)d_in[38];
    const float* logits_w = (const float*)d_in[39];
    const float* logits_b = (const float*)d_in[40];
    const float* delta_w = (const float*)d_in[41];
    const float* delta_b = (const float*)d_in[42];

    float* ws = (float*)d_ws;
    size_t off = 0;
    auto alloc = [&](size_t nfl) { float* p = ws + off; off += nfl; return p; };
    float* ste     = alloc((size_t)NN * 4 * DD);
    float* boxes   = alloc((size_t)NB * 2);
    float* pro0    = alloc((size_t)NB * DD);
    float* objA    = alloc((size_t)NB * DD);
    float* objB    = alloc((size_t)NB * DD);
    float* featb   = alloc((size_t)NB * RES_ * DD);
    float* qkvb    = alloc((size_t)NB * 3 * DD);
    float* attb    = alloc((size_t)NB * DD);
    float* pfb     = alloc((size_t)NB * DD);
    float* paramsb = alloc((size_t)NB * NP2);
    float* feat2b  = alloc((size_t)NB * RES_ * DD);
    float* tmp1    = alloc((size_t)NB * FF_);
    float* tmp2    = alloc((size_t)NB * FF_);
    float* ssb     = alloc((size_t)NN * 2 * DD);
    float* fcb     = alloc((size_t)NB * DD);
    float* cfb     = alloc((size_t)NB * DD);
    float* rfA     = alloc((size_t)NB * DD);
    float* rfB     = alloc((size_t)NB * DD);
    float* dlb     = alloc((size_t)NB * 2);

    float* logits_out = (float*)d_out;
    float* boxes_out  = (float*)d_out + (size_t)LL * NB * NC_;

    hipMemcpyAsync(boxes, bboxes, (size_t)NB * 2 * sizeof(float),
                   hipMemcpyDeviceToDevice, stream);

    te_kernel<<<1, 256, 0, stream>>>(t_in, tm_w1, tm_b1, tm_w2, tm_b2, ste);

    auto gemm = [&](const float* A, const float* W, const float* bias, float* C,
                    int M, int Ncols, int K, int relu) {
        dim3 g((Ncols + 31) / 32, (M + 31) / 32);
        gemm_f16_wmma<<<g, 32, 0, stream>>>(A, W, bias, C, M, Ncols, K, relu);
    };

    for (int i = 0; i < LL; ++i) {
        const float* pro_in = (i == 0) ? pro0 : ((i & 1) ? objA : objB);
        float* obj_out = (i & 1) ? objB : objA;

        // ROI sampling (+ pro on layer 0)
        roi_kernel<<<NB, 256, 0, stream>>>(features, boxes, featb, pro0, i == 0 ? 1 : 0);

        // Self-attention
        gemm(pro_in, attn_in_w + (size_t)i * 3 * DD * DD, attn_in_b + (size_t)i * 3 * DD,
             qkvb, NB, 3 * DD, DD, 0);
        attn_kernel<<<NN * NHH, 128, 0, stream>>>(qkvb, attb);
        gemm(attb, attn_out_w + (size_t)i * DD * DD, attn_out_b + (size_t)i * DD,
             tmp1, NB, DD, DD, 0);
        ln_kernel<<<NB, 256, 0, stream>>>(tmp1, pro_in, ln1_g + i * DD, ln1_b + i * DD,
                                          pfb, DD, 0);

        // Dynamic conv parameters (dominant GEMM: 400x256 @ 256x32768)
        gemm(pfb, dyn_w + (size_t)i * NP2 * DD, dyn_b + (size_t)i * NP2,
             paramsb, NB, NP2, DD, 0);
        dynconv_kernel<<<NB, 256, 0, stream>>>(featb, paramsb,
                                               dc_ln1_g + i * DYN_, dc_ln1_b + i * DYN_,
                                               dc_ln2_g + i * DD,   dc_ln2_b + i * DD,
                                               feat2b);
        gemm(feat2b, dc_out_w + (size_t)i * DD * RES_ * DD, dc_out_b + (size_t)i * DD,
             tmp1, NB, DD, RES_ * DD, 0);
        ln_kernel<<<NB, 256, 0, stream>>>(tmp1, nullptr, dc_ln3_g + i * DD, dc_ln3_b + i * DD,
                                          tmp2, DD, 1);
        ln_kernel<<<NB, 256, 0, stream>>>(tmp2, pfb, ln2_g + i * DD, ln2_b + i * DD,
                                          obj_out, DD, 0);

        // FFN
        gemm(obj_out, ffn_w1 + (size_t)i * FF_ * DD, ffn_b1 + (size_t)i * FF_,
             tmp1, NB, FF_, DD, 1);
        gemm(tmp1, ffn_w2 + (size_t)i * DD * FF_, ffn_b2 + (size_t)i * DD,
             tmp2, NB, DD, FF_, 0);
        ln_kernel<<<NB, 256, 0, stream>>>(tmp2, obj_out, ln3_g + i * DD, ln3_b + i * DD,
                                          obj_out, DD, 0);

        // Time FiLM
        gemm(ste, tb_w + (size_t)i * 2 * DD * 4 * DD, tb_b + (size_t)i * 2 * DD,
             ssb, NN, 2 * DD, 4 * DD, 0);
        film_kernel<<<(NB * DD + 255) / 256, 256, 0, stream>>>(obj_out, ssb, fcb);

        // Class head
        gemm(fcb, cls_w + (size_t)i * DD * DD, nullptr, tmp1, NB, DD, DD, 0);
        ln_kernel<<<NB, 256, 0, stream>>>(tmp1, nullptr, cls_ln_g + i * DD, cls_ln_b + i * DD,
                                          cfb, DD, 1);

        // Regression tower
        const float* rf_in = fcb;
        for (int j = 0; j < 3; ++j) {
            float* rf_out = (j & 1) ? rfB : rfA;
            gemm(rf_in, reg_w + ((size_t)i * 3 + j) * DD * DD, nullptr, tmp1, NB, DD, DD, 0);
            ln_kernel<<<NB, 256, 0, stream>>>(tmp1, nullptr,
                                              reg_ln_g + (i * 3 + j) * DD,
                                              reg_ln_b + (i * 3 + j) * DD,
                                              rf_out, DD, 1);
            rf_in = rf_out;
        }

        // Outputs
        gemm(cfb, logits_w + (size_t)i * NC_ * DD, logits_b + (size_t)i * NC_,
             logits_out + (size_t)i * NB * NC_, NB, NC_, DD, 0);
        gemm(rf_in, delta_w + (size_t)i * 2 * DD, delta_b + (size_t)i * 2,
             dlb, NB, 2, DD, 0);
        boxupdate_kernel<<<(NB + 255) / 256, 256, 0, stream>>>(dlb, boxes,
                                                               boxes_out + (size_t)i * NB * 2);
    }
}
